// ModConv_38543036514873
// MI455X (gfx1250) — matrix operand order repaired and checked by reading, not compiled, not run on hardware
//
#include <hip/hip_runtime.h>
#include <hip/hip_bf16.h>

// ---------------------------------------------------------------------------
// StyleGAN2 modulated conv as implicit GEMM on CDNA5 WMMA (gfx1250, wave32).
//   y[b,o,h,w] = sum_{c,kh,kw} x[b,c,h+kh-1,w+kw-1] * Wd[b,o,c,kh,kw]
// Conv = 9 shifted GEMMs of M=128(out ch) x K=128(in ch) x N=HW per batch,
// f16 inputs, f32 accumulation via v_wmma_f32_16x16x32_f16.
// x is pre-transposed to f16 channel-innermost so the LDS halo tile is a
// contiguous byte copy done with GLOBAL_LOAD_ASYNC_TO_LDS_B128 (ASYNCcnt).
// ---------------------------------------------------------------------------

#define B_SZ   16
#define CIN    128
#define COUT   128
#define HDIM   128
#define WDIM   128
#define WSTYLE 512
#define TILE_W 64          // pixels per workgroup tile (one row segment)
#define LDS_COLS 66        // TILE_W + 2 halo columns
#define LDS_CSTR 136       // padded channel stride (halves) to spread LDS banks

typedef _Float16 h16;
typedef __attribute__((ext_vector_type(8)))  _Float16 v8h;
typedef __attribute__((ext_vector_type(16))) _Float16 v16h;
typedef __attribute__((ext_vector_type(8)))  float    v8f;

union V16 { v16h v; v8h h[2]; };

// ---------------------------------------------------------------------------
// Kernel 1: style s[b,c] = dot(w[b,:], mod_w[c,:]) + mod_b[c] + 1
// ---------------------------------------------------------------------------
__global__ __launch_bounds__(128) void modconv_style(
    const float* __restrict__ w, const float* __restrict__ mod_w,
    const float* __restrict__ mod_b, float* __restrict__ s)
{
    const int b = blockIdx.x;
    const int c = threadIdx.x;
    const float* wb = w + (size_t)b * WSTYLE;
    const float* mw = mod_w + (size_t)c * WSTYLE;
    float acc = 0.0f;
    for (int j = 0; j < WSTYLE; j += 4) {
        acc += wb[j]   * mw[j]   + wb[j+1] * mw[j+1]
             + wb[j+2] * mw[j+2] + wb[j+3] * mw[j+3];
    }
    s[b * CIN + c] = acc + mod_b[c] + 1.0f;
}

// ---------------------------------------------------------------------------
// Kernel 2: modulate + demodulate, pack per-batch f16 weights:
//   Apack[b][tap][o][c] = f16( weight[o,c,tap] * s[b,c] * d[b,o] )
// ---------------------------------------------------------------------------
__global__ __launch_bounds__(128) void modconv_pack(
    const float* __restrict__ weight, const float* __restrict__ s,
    h16* __restrict__ Apack)
{
    __shared__ float red[CIN];
    const int b = blockIdx.x >> 7;      // / COUT
    const int o = blockIdx.x & (COUT - 1);
    const int c = threadIdx.x;

    const float sc = s[b * CIN + c];
    const float* wp = weight + ((size_t)o * CIN + c) * 9;
    float wk[9];
    float sum = 0.0f;
#pragma unroll
    for (int k = 0; k < 9; ++k) { wk[k] = wp[k] * sc; sum += wk[k] * wk[k]; }

    red[c] = sum;
    __syncthreads();
    for (int st = 64; st > 0; st >>= 1) {
        if (c < st) red[c] += red[c + st];
        __syncthreads();
    }
    const float d = __frsqrt_rn(red[0] + 1e-8f);

#pragma unroll
    for (int k = 0; k < 9; ++k)
        Apack[(((size_t)b * 9 + k) * COUT + o) * CIN + c] = (h16)(wk[k] * d);
}

// ---------------------------------------------------------------------------
// Kernel 3: transpose + convert x to f16 channel-innermost:
//   xT[b][h][w][c] = f16( x[b][c][h][w] )
// Block handles (b, h, 32-wide w tile) through an LDS transpose tile.
// ---------------------------------------------------------------------------
__global__ __launch_bounds__(256) void modconv_xpack(
    const float* __restrict__ x, h16* __restrict__ xT)
{
    __shared__ h16 t[CIN][33];          // +1 pad against bank conflicts
    const int b  = blockIdx.z;
    const int h  = blockIdx.y;
    const int w0 = blockIdx.x * 32;
    const int tid = threadIdx.x;

    // read phase: consecutive tid -> consecutive w (coalesced f32 reads)
    const int wi = tid & 31;
    const int cr = tid >> 5;            // 8 channel rows per iteration
#pragma unroll
    for (int it = 0; it < CIN / 8; ++it) {
        const int c = it * 8 + cr;
        t[c][wi] = (h16)x[(((size_t)b * CIN + c) * HDIM + h) * WDIM + w0 + wi];
    }
    __syncthreads();

    // write phase: consecutive tid -> consecutive c (coalesced f16 writes)
    const int ci = tid & 127;
    const int ws = tid >> 7;            // 2 w values per iteration
#pragma unroll
    for (int it = 0; it < 16; ++it) {
        const int wl = it * 2 + ws;
        xT[(((size_t)b * HDIM + h) * WDIM + w0 + wl) * CIN + ci] = t[ci][wl];
    }
}

// ---------------------------------------------------------------------------
// Shared WMMA compute + store: tap loop (9) x K chunks (4x32) -> 144 wmma/wave,
// then scatter C tiles per the f32 16x16 C/D VGPR layout.
// ---------------------------------------------------------------------------
__device__ __forceinline__ void gemm_compute_store(
    const h16* xs, const h16* __restrict__ Abase,
    int b, int h, int w0, int tid, float* __restrict__ out)
{
    const int lane = tid & 31;
    const int wave = tid >> 5;
    const int o0   = wave * 16;
    const int nrow = lane & 15;   // M (for A) / N (for B,C) index within tile
    const int half = lane >> 4;   // lane group 0/1 per ISA layouts

    v8f acc[4] = {};              // 4 x (16x16 f32) C tiles, 64 pixels

    for (int tap = 0; tap < 9; ++tap) {
        const int kh = tap / 3, kw = tap % 3;
        const h16* Arow = Abase + ((size_t)tap * COUT + (o0 + nrow)) * CIN;
        __builtin_prefetch(Arow + (size_t)COUT * CIN, 0, 1); // next tap's A
        const h16* xrow = xs + (kh * LDS_COLS + kw) * LDS_CSTR;

#pragma unroll
        for (int kc = 0; kc < CIN; kc += 32) {
            // A fragment, 16-bit A 16x32 layout (ISA 7.12.2)
            V16 a;
            a.h[0] = *(const v8h*)(Arow + kc + half * 8);
            a.h[1] = *(const v8h*)(Arow + kc + 16 + half * 8);

#pragma unroll
            for (int nt = 0; nt < 4; ++nt) {
                // B fragment: lane group holds K = 8*half+0..7 and 16+8*half+0..7 for N = nrow
                const h16* bp = xrow + (size_t)(nt * 16 + nrow) * LDS_CSTR
                                + kc + half * 8;
                V16 bf;
                bf.h[0] = *(const v8h*)(bp);
                bf.h[1] = *(const v8h*)(bp + 16);

                acc[nt] = __builtin_amdgcn_wmma_f32_16x16x32_f16(
                    false, a.v, false, bf.v, (short)0, acc[nt], false, false);
            }
        }
    }

    // store: C/D layout: elem r -> M = r + 8*half, N = nrow
    const size_t obase =
        (((size_t)b * COUT + o0 + half * 8) * HDIM + h) * WDIM + w0 + nrow;
#pragma unroll
    for (int nt = 0; nt < 4; ++nt) {
#pragma unroll
        for (int r = 0; r < 8; ++r) {
            out[obase + (size_t)r * HDIM * WDIM + nt * 16] = acc[nt][r];
        }
    }
}

// ---------------------------------------------------------------------------
// Kernel 4a: implicit-GEMM conv, async LDS staging path.
// Halo tile [3 rows][66 cols][128 ch] f16 copied with
// global_load_async_to_lds_b128; out-of-image cells zero-filled via LDS.
// ---------------------------------------------------------------------------
__global__ __launch_bounds__(256) void modconv_gemm_async(
    const h16* __restrict__ xT, const h16* __restrict__ Apack,
    float* __restrict__ out)
{
    __shared__ __align__(16) h16 xs[3 * LDS_COLS * LDS_CSTR];

    const int b   = blockIdx.y;
    const int h   = blockIdx.x >> 1;
    const int w0  = (blockIdx.x & 1) * TILE_W;
    const int tid = threadIdx.x;

    const v8h zero = {};
    const int NCHUNK = 3 * LDS_COLS * (CIN / 8);   // 16-byte chunks
    for (int idx = tid; idx < NCHUNK; idx += 256) {
        const int ch   = idx & 15;                 // chunk within 128-ch cell
        const int cell = idx >> 4;
        const int col  = cell % LDS_COLS;
        const int row  = cell / LDS_COLS;
        const int gh = h - 1 + row;
        const int gw = w0 - 1 + col;
        h16* dst = xs + (size_t)(row * LDS_COLS + col) * LDS_CSTR + ch * 8;
        if ((unsigned)gh < (unsigned)HDIM && (unsigned)gw < (unsigned)WDIM) {
            const h16* src =
                xT + (((size_t)b * HDIM + gh) * WDIM + gw) * CIN + ch * 8;
            // low 32 bits of a generic LDS pointer == wave-relative LDS addr
            unsigned ldsoff = (unsigned)(unsigned long long)dst;
            asm volatile("global_load_async_to_lds_b128 %0, %1, off"
                         :: "v"(ldsoff), "v"(src) : "memory");
        } else {
            *(v8h*)dst = zero;                     // zero-pad halo
        }
    }
    asm volatile("s_wait_asynccnt 0x0" ::: "memory");
    __syncthreads();

    gemm_compute_store(xs, Apack + (size_t)b * 9 * COUT * CIN,
                       b, h, w0, tid, out);
}

// ---------------------------------------------------------------------------
// Kernel 4b: fallback path (d_ws too small for xT): stage f32 x directly.
// ---------------------------------------------------------------------------
__global__ __launch_bounds__(256) void modconv_gemm_direct(
    const float* __restrict__ x, const h16* __restrict__ Apack,
    float* __restrict__ out)
{
    __shared__ __align__(16) h16 xs[3 * LDS_COLS * LDS_CSTR];

    const int b   = blockIdx.y;
    const int h   = blockIdx.x >> 1;
    const int w0  = (blockIdx.x & 1) * TILE_W;
    const int tid = threadIdx.x;

    const int NELEM = 3 * LDS_COLS * CIN;
    for (int idx = tid; idx < NELEM; idx += 256) {
        const int col = idx % LDS_COLS;
        const int t   = idx / LDS_COLS;
        const int c   = t % CIN;
        const int row = t / CIN;
        const int gh = h - 1 + row;
        const int gw = w0 - 1 + col;
        float v = 0.0f;
        if ((unsigned)gh < (unsigned)HDIM && (unsigned)gw < (unsigned)WDIM)
            v = x[(((size_t)b * CIN + c) * HDIM + gh) * WDIM + gw];
        xs[(row * LDS_COLS + col) * LDS_CSTR + c] = (h16)v;
    }
    __syncthreads();

    gemm_compute_store(xs, Apack + (size_t)b * 9 * COUT * CIN,
                       b, h, w0, tid, out);
}

// ---------------------------------------------------------------------------
extern "C" void kernel_launch(void* const* d_in, const int* in_sizes, int n_in,
                              void* d_out, int out_size, void* d_ws, size_t ws_size,
                              hipStream_t stream) {
    const float* x      = (const float*)d_in[0];
    const float* w      = (const float*)d_in[1];
    const float* weight = (const float*)d_in[2];
    const float* mod_w  = (const float*)d_in[3];
    const float* mod_b  = (const float*)d_in[4];
    float* out = (float*)d_out;

    const size_t apack_bytes = (size_t)B_SZ * 9 * COUT * CIN * sizeof(h16);
    const size_t s_bytes     = (size_t)B_SZ * CIN * sizeof(float);
    const size_t xT_bytes    = (size_t)B_SZ * HDIM * WDIM * CIN * sizeof(h16);

    h16*   Apack = (h16*)d_ws;
    float* s     = (float*)((char*)d_ws + apack_bytes);
    h16*   xT    = (h16*)((char*)d_ws + apack_bytes + s_bytes);

    modconv_style<<<dim3(B_SZ), 128, 0, stream>>>(w, mod_w, mod_b, s);
    modconv_pack<<<dim3(B_SZ * COUT), 128, 0, stream>>>(weight, s, Apack);

    if (ws_size >= apack_bytes + s_bytes + xT_bytes) {
        modconv_xpack<<<dim3(WDIM / 32, HDIM, B_SZ), 256, 0, stream>>>(x, xT);
        modconv_gemm_async<<<dim3(HDIM * (WDIM / TILE_W), B_SZ), 256, 0,
                             stream>>>(xT, Apack, out);
    } else {
        modconv_gemm_direct<<<dim3(HDIM * (WDIM / TILE_W), B_SZ), 256, 0,
                              stream>>>(x, Apack, out);
    }
}
